// FuzzyNeuron_89601607729513
// MI455X (gfx1250) — matrix operand order repaired and checked by reading, not compiled, not run on hardware
//
#include <hip/hip_runtime.h>

typedef __attribute__((ext_vector_type(16))) _Float16 v16h;
typedef __attribute__((ext_vector_type(8)))  float    v8f;

#define KTOT 131072
#define RDIM 64
#define ADIM 32
#define LOG2E 1.4426950408889634f

__global__ __launch_bounds__(256) void FuzzyNeuron_wmma_kernel(
    const float* __restrict__ input,   // [K, 32]
    const float* __restrict__ mu,      // [64, 32]
    const float* __restrict__ sigma,   // [64, 32]
    const float* __restrict__ rho,     // [64, 33]
    float* __restrict__ out)           // [K]
{
    // B fragments pre-packed per lane in contiguous k order (k = 16*hl + s, as A): chunk [a>>4][r], element a&15.
    // One chunk = one lane's 32-byte B-operand slice (2x ds_load_b128).
    // Expo coefficients are pre-scaled by log2(e) so the epilogue uses native exp2.
    __shared__ v16h  s_rho_h[2 * RDIM], s_rho_l[2 * RDIM];   // rho[r][a]^T
    __shared__ v16h  s_c_h  [2 * RDIM], s_c_l  [2 * RDIM];   // log2e/(2*sigma^2)
    __shared__ v16h  s_b_h  [2 * RDIM], s_b_l  [2 * RDIM];   // -2*mu*c*log2e
    __shared__ float s_bias [RDIM];                          // rho[r][32]
    __shared__ float s_const[RDIM];                          // log2e*sum_a mu^2*c

    #pragma unroll 2
    for (int idx = threadIdx.x; idx < ADIM * RDIM; idx += 256) {
        int r = idx >> 5, a = idx & 31;            // consecutive tid -> coalesced a
        float sg = sigma[r * ADIM + a];
        float cc = LOG2E / (2.0f * sg * sg);       // log2-domain coefficient
        float mm = mu[r * ADIM + a];
        float rr = rho[r * (ADIM + 1) + a];
        float bb = -2.0f * mm * cc;
        int   pe = ((a >> 4) * RDIM + r) * 16 + (a & 15);   // f16 slot
        _Float16 h;
        h = (_Float16)cc; ((_Float16*)s_c_h)[pe] = h;
        ((_Float16*)s_c_l)[pe] = (_Float16)(cc - (float)h);
        h = (_Float16)bb; ((_Float16*)s_b_h)[pe] = h;
        ((_Float16*)s_b_l)[pe] = (_Float16)(bb - (float)h);
        h = (_Float16)rr; ((_Float16*)s_rho_h)[pe] = h;
        ((_Float16*)s_rho_l)[pe] = (_Float16)(rr - (float)h);
    }
    if (threadIdx.x < RDIM) {
        int r = threadIdx.x;
        s_bias[r] = rho[r * (ADIM + 1) + ADIM];
        float acc = 0.0f;
        #pragma unroll
        for (int a = 0; a < ADIM; ++a) {
            float mm = mu[r * ADIM + a];
            float sg = sigma[r * ADIM + a];
            acc += (mm * mm) / (2.0f * sg * sg);
        }
        s_const[r] = acc * LOG2E;
    }
    __syncthreads();

    const int wave = threadIdx.x >> 5;
    const int lane = threadIdx.x & 31;
    const int m    = lane & 15;   // A row / B+C column index
    const int hl   = lane >> 4;   // half-wave select
    const int kblk = blockIdx.x * 256 + wave * 32;   // 2 k-tiles of 16 rows per wave

    for (int tt = 0; tt < 2; ++tt) {
        const int kt = kblk + tt * 16;
        __builtin_amdgcn_sched_barrier(0);   // no cross-tile pipelining (VGPR cap)

        // ---- A fragments: lane holds row kt+m, a in [16hl,16hl+16)
        const float* xrow = input + (size_t)(kt + m) * ADIM + 16 * hl;
        float4 f0 = *(const float4*)(xrow);
        float4 f1 = *(const float4*)(xrow + 4);
        float4 f2 = *(const float4*)(xrow + 8);
        float4 f3 = *(const float4*)(xrow + 12);
        float xa[16] = { f0.x, f0.y, f0.z, f0.w,  f1.x, f1.y, f1.z, f1.w,
                         f2.x, f2.y, f2.z, f2.w,  f3.x, f3.y, f3.z, f3.w };
        float sq[16];
        #pragma unroll
        for (int e = 0; e < 16; ++e) sq[e] = xa[e] * xa[e];   // v_pk_mul_f32 pairs

        // Hi parts (v_cvt_pk_f16_f32 pairs); vector elements are the ONLY
        // definition of hi. Both residual loops have the identical
        // "val - (float)hi" shape so they lower as v_fma_mix_f32 reading the
        // packed f16 halves directly (no duplicate or back converts).
        v16h Axh, Axl, Ash, Asl;
        #pragma unroll
        for (int e = 0; e < 16; ++e) {
            Axh[e] = (_Float16)xa[e];
            Ash[e] = (_Float16)sq[e];
        }
        #pragma unroll
        for (int e = 0; e < 16; ++e) {
            Axl[e] = (_Float16)(xa[e] - (float)Axh[e]);
            Asl[e] = (_Float16)(sq[e] - (float)Ash[e]);
        }

        v8f zacc[4] = {};   // z GEMM
        v8f eacc[4] = {};   // expo GEMMs in log2 domain (x^2*c + x*(-2 mu c))

        #pragma unroll
        for (int t = 0; t < 4; ++t) {
            const int ch = hl * RDIM + t * 16 + m;   // B chunk for this lane/tile
            v16h Brh = s_rho_h[ch], Brl = s_rho_l[ch];
            zacc[t] = __builtin_amdgcn_wmma_f32_16x16x32_f16(
                false, Axh, false, Brh, (short)0, zacc[t], false, false);
            zacc[t] = __builtin_amdgcn_wmma_f32_16x16x32_f16(
                false, Axh, false, Brl, (short)0, zacc[t], false, false);
            zacc[t] = __builtin_amdgcn_wmma_f32_16x16x32_f16(
                false, Axl, false, Brh, (short)0, zacc[t], false, false);

            v16h Bch = s_c_h[ch], Bcl = s_c_l[ch];
            eacc[t] = __builtin_amdgcn_wmma_f32_16x16x32_f16(
                false, Ash, false, Bch, (short)0, eacc[t], false, false);
            eacc[t] = __builtin_amdgcn_wmma_f32_16x16x32_f16(
                false, Ash, false, Bcl, (short)0, eacc[t], false, false);
            eacc[t] = __builtin_amdgcn_wmma_f32_16x16x32_f16(
                false, Asl, false, Bch, (short)0, eacc[t], false, false);

            v16h Bbh = s_b_h[ch], Bbl = s_b_l[ch];
            eacc[t] = __builtin_amdgcn_wmma_f32_16x16x32_f16(
                false, Axh, false, Bbh, (short)0, eacc[t], false, false);
            eacc[t] = __builtin_amdgcn_wmma_f32_16x16x32_f16(
                false, Axh, false, Bbl, (short)0, eacc[t], false, false);
            eacc[t] = __builtin_amdgcn_wmma_f32_16x16x32_f16(
                false, Axl, false, Bbh, (short)0, eacc[t], false, false);

            __builtin_amdgcn_sched_barrier(0);   // keep only this tile's B frags live
        }

        // ---- Epilogue: lane L, VGPR v holds (row = v + 8*hl, r = 16*t + m).
        float* outp = out + kt;
        #pragma unroll
        for (int v = 0; v < 8; ++v) {
            float num = 0.0f, den = 0.0f;
            #pragma unroll
            for (int t = 0; t < 4; ++t) {
                int r = t * 16 + m;
                float z  = zacc[t][v] + s_bias[r];
                float ex = eacc[t][v] + s_const[r];
                float w  = __builtin_amdgcn_exp2f(-ex);   // v_exp_f32, neg folds
                num = fmaf(z, w, num);
                den += w;
            }
            #pragma unroll
            for (int off = 1; off < 16; off <<= 1) {
                num += __shfl_xor(num, off, 32);
                den += __shfl_xor(den, off, 32);
            }
            if (m == 0)
                outp[v + 8 * hl] = num / (den + 1e-13f);
        }
        __builtin_amdgcn_sched_barrier(0);
    }
}

extern "C" void kernel_launch(void* const* d_in, const int* in_sizes, int n_in,
                              void* d_out, int out_size, void* d_ws, size_t ws_size,
                              hipStream_t stream) {
    const float* input = (const float*)d_in[0];
    const float* mu    = (const float*)d_in[1];
    const float* sigma = (const float*)d_in[2];
    const float* rho   = (const float*)d_in[3];
    float* out = (float*)d_out;

    dim3 block(256);
    dim3 grid(KTOT / 256);   // 8 waves/block, 2x16 k-rows per wave
    FuzzyNeuron_wmma_kernel<<<grid, block, 0, stream>>>(input, mu, sigma, rho, out);
}
